// CrossAttentionBlock_41102837022983
// MI455X (gfx1250) — matrix operand order repaired and checked by reading, not compiled, not run on hardware
//
#include <hip/hip_runtime.h>

// ---------------------------------------------------------------------------
// CrossAttentionBlock for MI455X (gfx1250, wave32, WMMA f16 -> f32 accum,
// TDM async tensor loads for K/V staging).
// B=8, NQ=NKV=4096, QD=KVD=1024, AD=128
// ---------------------------------------------------------------------------

typedef __attribute__((ext_vector_type(16))) _Float16 v16h;
typedef __attribute__((ext_vector_type(8)))  _Float16 v8h;
typedef __attribute__((ext_vector_type(8)))  float    v8f;
typedef unsigned int u32x4 __attribute__((ext_vector_type(4)));
typedef int          i32x4 __attribute__((ext_vector_type(4)));
typedef int          i32x8 __attribute__((ext_vector_type(8)));

#define CB_B    8
#define CB_NQ   4096
#define CB_NKV  4096
#define CB_QD   1024
#define CB_AD   128
#define KVSTEP  64

// D = A*B + C, 16x16x32 f16 (wave32)
__device__ __forceinline__ v8f wmma_f16(v16h a, v16h b, v8f c) {
  return __builtin_amdgcn_wmma_f32_16x16x32_f16(
      /*neg_a=*/false, a, /*neg_b=*/false, b,
      /*c_mod=*/(short)0, c, /*reuse_a=*/false, /*reuse_b=*/false);
}

// A-fragment loader (16x32 f16, ISA 7.12.2): per lane, row = lane&15,
// elements e=0..7 <- k = base+0..7, e=8..15 <- k = base+16..23,
// base = (lane>=16 ? 8 : 0) (caller passes pointer at row+chunk+base).
__device__ __forceinline__ v16h ld_a_f16(const _Float16* p) {
  v8h lo = *reinterpret_cast<const v8h*>(p);
  v8h hi = *reinterpret_cast<const v8h*>(p + 16);
  return __builtin_shufflevector(lo, hi, 0, 1, 2, 3, 4, 5, 6, 7,
                                         8, 9, 10, 11, 12, 13, 14, 15);
}

// A-fragment from fp32 source with on-the-fly f16 conversion.
__device__ __forceinline__ v16h ld_a_f32(const float* p) {
  v16h r;
#pragma unroll
  for (int i = 0; i < 8; ++i) {
    r[i]     = (_Float16)p[i];
    r[8 + i] = (_Float16)p[16 + i];
  }
  return r;
}

// B-fragment loader (32x16 f16): per lane, column n = lane&15,
// e=0..7 <- k = base+0..7, e=8..15 <- k = base+16..23,
// base = (lane>=16 ? 8 : 0) (caller applies koff).
__device__ __forceinline__ v16h ld_b_f16(const _Float16* p) {
  v8h lo = *reinterpret_cast<const v8h*>(p);
  v8h hi = *reinterpret_cast<const v8h*>(p + 16);
  return __builtin_shufflevector(lo, hi, 0, 1, 2, 3, 4, 5, 6, 7,
                                         8, 9, 10, 11, 12, 13, 14, 15);
}

// half-wave (16 lane) reductions: C/D fragment rows live within a lane-half
__device__ __forceinline__ float half_max16(float x) {
#pragma unroll
  for (int off = 1; off < 16; off <<= 1)
    x = fmaxf(x, __shfl_xor(x, off, 32));
  return x;
}
__device__ __forceinline__ float half_sum16(float x) {
#pragma unroll
  for (int off = 1; off < 16; off <<= 1)
    x += __shfl_xor(x, off, 32);
  return x;
}

// LDS byte offset of a generic pointer known to point into LDS.
__device__ __forceinline__ unsigned int lds_offset(const void* p) {
  return (unsigned int)(unsigned long long)
      (const __attribute__((address_space(3))) char*)p;
}

// ---------------------------------------------------------------------------
// Tensor Data Mover: async 2D tile load Global -> LDS (f16 elements).
// D# per CDNA5 ISA §8: group0 = {flags, lds_addr, global_addr, type},
// group1 = {data_size, tensor dims, tile dims, dim0 stride}. Dims/strides in
// element (2-byte) units; tile stored contiguously in LDS.
// ---------------------------------------------------------------------------
__device__ __forceinline__ void tdm_load_2d_f16(
    const void* gptr, unsigned int lds_byte_off,
    unsigned int tile_d0, unsigned int tile_d1,
    unsigned int tensor_d0, unsigned int tensor_d1, unsigned int stride_d0) {
  unsigned long long ga = (unsigned long long)gptr;
  u32x4 g0;
  g0.x = 1u;                                   // count=1 (valid user D#)
  g0.y = lds_byte_off;                         // lds_addr (bytes)
  g0.z = (unsigned int)ga;                     // global_addr[31:0]
  g0.w = (unsigned int)((ga >> 32) & 0x1FFFFFFu) | (2u << 30);  // [56:32]|type=2
  i32x8 g1;
  g1[0] = (int)(1u << 16);                     // workgroup_mask=0, data_size=1 (2B)
  g1[1] = (int)((tensor_d0 & 0xFFFFu) << 16);  // tensor_dim0[15:0]
  g1[2] = (int)((tensor_d0 >> 16) | ((tensor_d1 & 0xFFFFu) << 16));
  g1[3] = (int)((tensor_d1 >> 16) | (tile_d0 << 16));
  g1[4] = (int)tile_d1;                        // tile_dim1 | tile_dim2(unused)=0
  g1[5] = (int)stride_d0;                      // tensor_dim0_stride[31:0]
  g1[6] = 0;
  g1[7] = 0;
  i32x4 z4 = {0, 0, 0, 0};
#if __clang_major__ >= 23
  i32x8 z8 = {0, 0, 0, 0, 0, 0, 0, 0};
  __builtin_amdgcn_tensor_load_to_lds(g0, g1, z4, z4, z8, 0);
#else
  __builtin_amdgcn_tensor_load_to_lds(g0, g1, z4, z4, 0);
#endif
}

// ---------------------------------------------------------------------------
// Kernel 0: W [QD][AD] f32 -> Wt [AD][QD] f16 (B-fragment friendly)
// ---------------------------------------------------------------------------
__global__ __launch_bounds__(256) void cab_wt_kernel(const float* __restrict__ W,
                                                     _Float16* __restrict__ Wt) {
  int idx = blockIdx.x * blockDim.x + threadIdx.x;  // < QD*AD = 131072
  int kk = idx >> 7;                                // / AD
  int nn = idx & (CB_AD - 1);
  Wt[(size_t)nn * CB_QD + kk] = (_Float16)W[idx];
}

// ---------------------------------------------------------------------------
// Kernel 1: projection GEMM. dst = (src[Mx1024] @ W[1024x128]) * scale, f16.
// One wave -> 16 rows x 128 cols. vtrans=1 stores V transposed per batch.
// ---------------------------------------------------------------------------
__global__ __launch_bounds__(128) void cab_proj_kernel(
    const float* __restrict__ src, const _Float16* __restrict__ Wt,
    _Float16* __restrict__ dst, float scale, int vtrans) {
  const int lane  = threadIdx.x & 31;
  const int wave  = blockIdx.x * (blockDim.x >> 5) + (threadIdx.x >> 5);
  const int row0  = wave << 4;
  const int arow  = lane & 15;
  const int abase = (lane >> 4) << 3;      // 0 or 8   (A frag k-base)
  const int koff  = (lane >> 4) << 3;      // 0 or 8   (B frag k-base)
  const int bn    = lane & 15;             // B/C/D column
  const int crow0 = (lane >> 4) << 3;      // C/D row base: 0 or 8

  const float* rowp = src + (size_t)(row0 + arow) * CB_QD;

  v8f acc[8];
#pragma unroll
  for (int nt = 0; nt < 8; ++nt) acc[nt] = (v8f)(0.0f);

  for (int c = 0; c < CB_QD; c += 32) {
    __builtin_prefetch(rowp + c + 128, 0, 3);   // global_prefetch_b8
    v16h a = ld_a_f32(rowp + c + abase);
#pragma unroll
    for (int nt = 0; nt < 8; ++nt) {
      v16h b = ld_b_f16(Wt + (size_t)(nt * 16 + bn) * CB_QD + c + koff);
      acc[nt] = wmma_f16(a, b, acc[nt]);
    }
  }

#pragma unroll
  for (int nt = 0; nt < 8; ++nt) {
#pragma unroll
    for (int r = 0; r < 8; ++r) {
      float val = acc[nt][r] * scale;
      int grow = row0 + crow0 + r;
      int gcol = nt * 16 + bn;
      if (!vtrans) {
        dst[(size_t)grow * CB_AD + gcol] = (_Float16)val;
      } else {
        int bb = grow >> 12;
        int mm = grow & (CB_NKV - 1);
        dst[((size_t)bb * CB_AD + gcol) * CB_NKV + mm] = (_Float16)val;
      }
    }
  }
}

// ---------------------------------------------------------------------------
// Kernel 2: flash attention with TDM-staged K/V tiles.
// Block = 4 waves = 64 q rows of one batch. Per 64-column KV step:
// wave0 DMAs next K (64x128) and V^T (128x64) tiles into the alternate LDS
// buffer (tensor_load_to_lds), everyone computes from the current buffer:
// 16 WMMA for S, online softmax, P through per-wave LDS, 16 WMMA for O+=P*V.
// ---------------------------------------------------------------------------
__global__ __launch_bounds__(128) void cab_attn_kernel(
    const _Float16* __restrict__ Qp,   // [B*NQ][AD] f16 (scale folded in)
    const _Float16* __restrict__ Kp,   // [B*NKV][AD] f16
    const _Float16* __restrict__ Vt,   // [B][AD][NKV] f16
    float* __restrict__ out) {         // [B][NQ][AD] f32
  __shared__ __align__(16) _Float16 kts[2][KVSTEP * CB_AD];   // 2 x 16KB
  __shared__ __align__(16) _Float16 vts[2][CB_AD * KVSTEP];   // 2 x 16KB
  __shared__ __align__(16) _Float16 plds[4][16 * KVSTEP];     // 8KB

  const int warp  = threadIdx.x >> 5;
  const int lane  = threadIdx.x & 31;
  const int wave  = blockIdx.x * 4 + warp;
  const int bq    = wave >> 8;          // batch (256 q-tiles per batch)
  const int q0    = (wave & 255) << 4;  // q row base
  const int arow  = lane & 15;
  const int abase = (lane >> 4) << 3;   // 0 or 8
  const int koff  = (lane >> 4) << 3;   // 0 or 8
  const int bn    = lane & 15;
  const int crow0 = (lane >> 4) << 3;   // 0 or 8

  // Resident Q A-fragments (AD=128 -> 4 chunks of 32)
  const _Float16* qrow = Qp + (size_t)(bq * CB_NQ + q0 + arow) * CB_AD;
  v16h qa[4];
#pragma unroll
  for (int c = 0; c < 4; ++c) qa[c] = ld_a_f16(qrow + c * 32 + abase);

  float m_i[8], l_i[8];
  v8f o[8];
#pragma unroll
  for (int r = 0; r < 8; ++r) { m_i[r] = -1e30f; l_i[r] = 0.0f; }
#pragma unroll
  for (int nt = 0; nt < 8; ++nt) o[nt] = (v8f)(0.0f);

  const _Float16* Kb  = Kp + (size_t)bq * CB_NKV * CB_AD;
  const _Float16* Vtb = Vt + (size_t)bq * CB_AD * CB_NKV;
  _Float16* pl = &plds[warp][0];

  const int NT = CB_NKV / KVSTEP;  // 64 iterations

  // Prologue: DMA tile 0 into buffer 0 (one wave drives the TDM).
  if (warp == 0) {
    tdm_load_2d_f16(Kb, lds_offset(&kts[0][0]),
                    CB_AD, KVSTEP, CB_AD, CB_NKV, CB_AD);
    tdm_load_2d_f16(Vtb, lds_offset(&vts[0][0]),
                    KVSTEP, CB_AD, CB_NKV, CB_AD, CB_NKV);
  }

  for (int it = 0; it < NT; ++it) {
    const int cur = it & 1;
    if (it + 1 < NT) {
      if (warp == 0) {
        const int nxt = (it + 1) & 1;
        tdm_load_2d_f16(Kb + (size_t)(it + 1) * KVSTEP * CB_AD,
                        lds_offset(&kts[nxt][0]),
                        CB_AD, KVSTEP, CB_AD, CB_NKV, CB_AD);
        tdm_load_2d_f16(Vtb + (size_t)(it + 1) * KVSTEP,
                        lds_offset(&vts[nxt][0]),
                        KVSTEP, CB_AD, CB_NKV, CB_AD, CB_NKV);
      }
      __builtin_amdgcn_s_wait_tensorcnt(2);  // tile `it` complete (in-order)
    } else {
      __builtin_amdgcn_s_wait_tensorcnt(0);
    }
    __syncthreads();  // LDS tile visible to all 4 waves

    const _Float16* kt = &kts[cur][0];
    const _Float16* vt = &vts[cur][0];

    // ---- S = Q @ K^T for 64 kv columns (four 16x16 C/D fragments) ----
    v8f s[4];
#pragma unroll
    for (int t = 0; t < 4; ++t) s[t] = (v8f)(0.0f);
#pragma unroll
    for (int j = 0; j < 2; ++j) {
#pragma unroll
      for (int c = 0; c < 4; ++c) {
        v16h kb0 = ld_b_f16(kt + (size_t)(j * 32 + bn)      * CB_AD + c * 32 + koff);
        v16h kb1 = ld_b_f16(kt + (size_t)(j * 32 + 16 + bn) * CB_AD + c * 32 + koff);
        s[2 * j]     = wmma_f16(qa[c], kb0, s[2 * j]);
        s[2 * j + 1] = wmma_f16(qa[c], kb1, s[2 * j + 1]);
      }
    }

    // ---- online softmax (scale already folded into Q) ----
#pragma unroll
    for (int r = 0; r < 8; ++r) {
      float mx = fmaxf(fmaxf(s[0][r], s[1][r]), fmaxf(s[2][r], s[3][r]));
      mx = half_max16(mx);
      float mnew  = fmaxf(m_i[r], mx);
      float alpha = __expf(m_i[r] - mnew);
      float p0 = __expf(s[0][r] - mnew);
      float p1 = __expf(s[1][r] - mnew);
      float p2 = __expf(s[2][r] - mnew);
      float p3 = __expf(s[3][r] - mnew);
      l_i[r] = l_i[r] * alpha + half_sum16((p0 + p1) + (p2 + p3));
      m_i[r] = mnew;
#pragma unroll
      for (int nt = 0; nt < 8; ++nt) o[nt][r] *= alpha;
      // P tile (16x64 f16) to per-wave LDS, C/D layout -> row major
      pl[(crow0 + r) * KVSTEP + bn]      = (_Float16)p0;
      pl[(crow0 + r) * KVSTEP + bn + 16] = (_Float16)p1;
      pl[(crow0 + r) * KVSTEP + bn + 32] = (_Float16)p2;
      pl[(crow0 + r) * KVSTEP + bn + 48] = (_Float16)p3;
    }

    // ---- O += P @ V (P reloaded as two 16x32 A fragments) ----
#pragma unroll
    for (int j = 0; j < 2; ++j) {
      v16h pa = ld_a_f16(pl + arow * KVSTEP + j * 32 + abase);
#pragma unroll
      for (int nt = 0; nt < 8; ++nt) {
        v16h vb = ld_b_f16(vt + (size_t)(nt * 16 + bn) * KVSTEP + j * 32 + koff);
        o[nt] = wmma_f16(pa, vb, o[nt]);
      }
    }
    __syncthreads();  // all reads of buffer `cur` done before its next DMA
  }

  // ---- epilogue: normalize and store fp32 ----
#pragma unroll
  for (int nt = 0; nt < 8; ++nt) {
#pragma unroll
    for (int r = 0; r < 8; ++r) {
      out[((size_t)bq * CB_NQ + q0 + crow0 + r) * CB_AD + nt * 16 + bn] =
          o[nt][r] / l_i[r];
    }
  }
}

// ---------------------------------------------------------------------------
extern "C" void kernel_launch(void* const* d_in, const int* in_sizes, int n_in,
                              void* d_out, int out_size, void* d_ws, size_t ws_size,
                              hipStream_t stream) {
  (void)in_sizes; (void)n_in; (void)out_size; (void)ws_size;
  const float* q  = (const float*)d_in[0];
  const float* k  = (const float*)d_in[1];
  const float* v  = (const float*)d_in[2];
  const float* Wq = (const float*)d_in[3];
  const float* Wk = (const float*)d_in[4];
  const float* Wv = (const float*)d_in[5];

  _Float16* ws  = (_Float16*)d_ws;
  const size_t WT_ELEMS = (size_t)CB_QD * CB_AD;          // 131072
  const size_t P_ELEMS  = (size_t)CB_B * CB_NQ * CB_AD;   // 4194304
  _Float16* Wqt = ws;
  _Float16* Wkt = Wqt + WT_ELEMS;
  _Float16* Wvt = Wkt + WT_ELEMS;
  _Float16* Qp  = Wvt + WT_ELEMS;
  _Float16* Kp  = Qp + P_ELEMS;
  _Float16* Vt  = Kp + P_ELEMS;

  // 1) weight transpose + f16 convert
  cab_wt_kernel<<<512, 256, 0, stream>>>(Wq, Wqt);
  cab_wt_kernel<<<512, 256, 0, stream>>>(Wk, Wkt);
  cab_wt_kernel<<<512, 256, 0, stream>>>(Wv, Wvt);

  // 2) projections (32768 rows, 16 rows/wave, 4 waves/block -> 512 blocks)
  const float s = 0.08838834764831845f;  // AD^-0.5, folded into Q
  cab_proj_kernel<<<512, 128, 0, stream>>>(q, Wqt, Qp, s, 0);
  cab_proj_kernel<<<512, 128, 0, stream>>>(k, Wkt, Kp, 1.0f, 0);
  cab_proj_kernel<<<512, 128, 0, stream>>>(v, Wvt, Vt, 1.0f, 1);

  // 3) flash attention (2048 waves / 4 per block -> 512 blocks)
  cab_attn_kernel<<<512, 128, 0, stream>>>(Qp, Kp, Vt, (float*)d_out);
}
